// Correspondence_Flow_Net_71390946394403
// MI455X (gfx1250) — matrix operand order repaired and checked by reading, not compiled, not run on hardware
//
#include <hip/hip_runtime.h>
#include <hip/hip_fp16.h>
#include <cstdint>

typedef _Float16 h16;
typedef __attribute__((ext_vector_type(16))) _Float16 v16h;
typedef __attribute__((ext_vector_type(4)))  _Float16 h4;
typedef __attribute__((ext_vector_type(8)))  float    v8f;
typedef __attribute__((ext_vector_type(4)))  unsigned v4u;
typedef __attribute__((ext_vector_type(8)))  int      v8i;
typedef __attribute__((ext_vector_type(4)))  int      v4i;

#define BATCH 4
#define HW256 256          // H == W == 256, compile-time (shifts, no divisions)
#define TW    64           // pixels per block along x for WMMA conv kernel

#if defined(__has_builtin)
#if __has_builtin(__builtin_amdgcn_tensor_load_to_lds) && __has_builtin(__builtin_amdgcn_s_wait_tensorcnt)
#define HAVE_TDM 1
#endif
#endif
#ifndef HAVE_TDM
#define HAVE_TDM 0
#endif

union VU { v16h v; uint4 q[2]; };

// ---------------------------------------------------------------------------
// TDM: DMA a [rows x wco] f16 tile of the [*][Co] weight matrix into LDS.
// D# packed per CDNA5 ISA 8.3/8.4: group0 {count=1, lds_addr, global_addr,
// type=2}; group1 {data_size=2B, tensor_dim0=wco, tensor_dim1=rows,
// tile_dim0=wco, tile_dim1=rows, tensor_dim0_stride=Co}. Groups 2/3 unused.
// ---------------------------------------------------------------------------
__device__ __forceinline__ void tdm_load_panel(const h16* gsrc, unsigned ldsoff,
                                               int rows, int wco, int Co)
{
#if HAVE_TDM
  unsigned long long ga = (unsigned long long)(uintptr_t)gsrc;
  v4u g0 = {1u,                                   // count=1 (valid descriptor)
            ldsoff,                               // LDS byte address
            (unsigned)ga,                         // global_addr[31:0]
            (unsigned)(ga >> 32) | 0x80000000u};  // global_addr[56:32] | type=2
  v8i g1 = {(int)(1u << 16),                      // data_size = 1 -> 2 bytes
            (int)((unsigned)wco << 16),           // tensor_dim0[15:0]
            (int)((unsigned)rows << 16),          // tensor_dim1[15:0]
            (int)((unsigned)wco << 16),           // tile_dim0
            rows,                                 // tile_dim1 (tile_dim2=0)
            Co,                                   // tensor_dim0_stride[31:0]
            0, 0};
  v4i z4 = {0, 0, 0, 0};
#if __clang_major__ >= 23
  v8i z8 = {0, 0, 0, 0, 0, 0, 0, 0};
  __builtin_amdgcn_tensor_load_to_lds(g0, g1, z4, z4, z8, 0);
#else
  __builtin_amdgcn_tensor_load_to_lds(g0, g1, z4, z4, 0);
#endif
#else
  (void)gsrc; (void)ldsoff; (void)rows; (void)wco; (void)Co;
#endif
}

__device__ __forceinline__ void tdm_wait(int n)
{
#if HAVE_TDM
  if (n) __builtin_amdgcn_s_wait_tensorcnt(1);
  else   __builtin_amdgcn_s_wait_tensorcnt(0);
#else
  (void)n;
#endif
}

// ---------------------------------------------------------------------------
// Fold inference BN into conv weights:  y = conv(x)*s + (b - m*s),
// s = g/sqrt(v+eps).  Output layout: f16 wout[(ky*KW+kx)*Ci + ci][co].
// ---------------------------------------------------------------------------
__global__ __launch_bounds__(256) void k_prep_w(
    const float* __restrict__ w, const float* __restrict__ g,
    const float* __restrict__ bb, const float* __restrict__ mm,
    const float* __restrict__ vv, h16* __restrict__ wout,
    float* __restrict__ bout, int Co, int Ci, int KH, int KW)
{
  int i = blockIdx.x * 256 + threadIdx.x;
  int K = Ci * KH * KW;
  if (i >= Co * K) return;
  int co = i / K, k = i - co * K;
  int ci = k % Ci, tap = k / Ci;
  int ky = tap / KW, kx = tap - ky * KW;
  float s = 1.f;
  if (g) s = g[co] * rsqrtf(vv[co] + 1e-5f);
  wout[(size_t)k * Co + co] =
      (h16)(w[(((size_t)co * Ci + ci) * KH + ky) * KW + kx] * s);
  if (k == 0) bout[co] = bb[co] - (mm ? mm[co] * s : 0.f);
}

// ---------------------------------------------------------------------------
// Stem conv: 7x7, Ci=1 -> 16ch, BN+ReLU folded (0.05% of FLOPs, direct conv).
// ---------------------------------------------------------------------------
__global__ __launch_bounds__(256) void k_conv_stem(
    const float* __restrict__ in, const h16* __restrict__ wgt,  // [tap][16]
    const float* __restrict__ bias, h16* __restrict__ out)
{
  __shared__ float s_p[22 * 22];
  const int t  = blockIdx.x;
  const int tx = t & 15, ty = (t >> 4) & 15, b = t >> 8;
  const int x0 = tx * 16, y0 = ty * 16;
  const int tid = threadIdx.x;

  for (int i = tid; i < 22 * 22; i += 256) {
    int r = i / 22, c = i - r * 22;
    int yy = y0 + r - 3, xx = x0 + c - 3;
    float v = 0.f;
    if ((unsigned)yy < 256u && (unsigned)xx < 256u)
      v = in[(((size_t)b << 8) + yy) * 256 + xx];
    s_p[i] = v;
  }
  __syncthreads();

  const int py = tid >> 4, px = tid & 15;
  float acc[16];
  #pragma unroll
  for (int c = 0; c < 16; ++c) acc[c] = bias[c];
  #pragma unroll 7
  for (int tap = 0; tap < 49; ++tap) {
    int ky = tap / 7, kx = tap - ky * 7;
    float v = s_p[(py + ky) * 22 + (px + kx)];
    const h16* wp = &wgt[tap * 16];
    #pragma unroll
    for (int c = 0; c < 16; ++c) acc[c] += v * (float)wp[c];
  }
  VU o;
  #pragma unroll
  for (int c = 0; c < 16; ++c) {
    float v = acc[c] > 0.f ? acc[c] : 0.f;
    o.v[c] = (h16)v;
  }
  *(v16h*)&out[((((size_t)b << 8) + y0 + py) * 256 + x0 + px) * 16] = o.v;
}

// ---------------------------------------------------------------------------
// Implicit-GEMM conv via V_WMMA_F32_16X16X32_F16, templated on shape so all
// per-chunk address math folds to constants.  H = W = 256 compile-time.
//   activations: NHWC f16;  weights: f16 [k = (ky*KW+kx)*CI + ci][co]
// Block: 256 thr = 8 waves; tile M=64 pixels (row y), N=32 co.
// Input patch staged in LDS; weight B-panel staged by the Tensor Data Mover
// in 64-K-row double-buffered stages (TDM overlaps WMMA compute).
// Per 32-K chunk: A = 2 x ds_load_b128 halves (each inside one tap),
// B = 16 strided f16 LDS reads, 1 WMMA.  Epilogue fuses bias/residual/ReLU.
// ---------------------------------------------------------------------------
template <int KH, int KW, int CI>
__global__ __launch_bounds__(256) void k_conv_wmma(
    const h16* __restrict__ in, const h16* __restrict__ wgt,
    const float* __restrict__ bias, const h16* __restrict__ skip,
    h16* __restrict__ out, int Co, int relu)
{
  constexpr int PW   = TW + KW - 1;
  constexpr int PAD  = KW / 2;
  constexpr int KTOT = KH * KW * CI;
  constexpr int L2CI = (CI == 16 ? 4 : CI == 32 ? 5 : CI == 64 ? 6 : 7);
  constexpr int KCH  = 64;                        // K rows per TDM stage
  constexpr int NSTG = (KTOT + KCH - 1) / KCH;

  __shared__ h16 s_in[KH * PW * CI];
  __shared__ h16 s_w[2][KCH * 32];                // double-buffered B panel

  const int tid  = threadIdx.x;
  const int lane = tid & 31;
  const int wv   = tid >> 5;
  const int mi   = wv & 3, ni = wv >> 2;
  const int by   = blockIdx.y;
  const int b    = by >> 8, y = by & 255;
  const int x0   = blockIdx.x * TW;
  const int co0  = (blockIdx.z * 2 + ni) * 16;
  const bool active = (co0 < Co);
  const int wcosh = (Co == 16) ? 4 : 5;           // LDS panel row stride shift
  const int wco   = 1 << wcosh;
  const h16* gpanel = &wgt[blockIdx.z * 32];      // tile origin in weight matrix

  // ---- stage 0 B-panel DMA (wave 0) ----
  if (wv == 0)
    tdm_load_panel(gpanel, (unsigned)(uintptr_t)&s_w[0][0],
                   KTOT < KCH ? KTOT : KCH, wco, Co);

  // ---- cooperative LDS stage of the input patch (zero-padded borders) ----
  constexpr int NVEC = (KH * PW * CI) / 8;
  for (int i = tid; i < NVEC; i += 256) {
    int e  = i << 3;
    int ci = e % CI;                  // multiple of 8, stays within one pixel
    int rest = e / CI;
    int xl = rest % PW, r = rest / PW;
    int yy = y + r - PAD, xx = x0 + xl - PAD;
    uint4 val = make_uint4(0u, 0u, 0u, 0u);
    if ((unsigned)yy < 256u && (unsigned)xx < 256u)
      val = *(const uint4*)&in[((((size_t)b << 8) + yy) * 256 + xx) * CI + ci];
    *(uint4*)&s_in[e] = val;
  }

  v8f acc = {0.f, 0.f, 0.f, 0.f, 0.f, 0.f, 0.f, 0.f};

  const int m     = lane & 15;
  const int khalf = (lane >> 4) << 3;             // A-layout lane half (0|8)
  const int xbase = mi * 16 + m;

  // one 16-K half of the A fragment: 8 contiguous f16 within one tap
  auto loadA = [&](int kb) -> uint4 {
    int tap = kb >> L2CI;
    int ci0 = (kb & (CI - 1)) + khalf;
    int ky  = tap / KW, kx = tap - ky * KW;       // template-constant div
    return *(const uint4*)&s_in[((ky * PW) + (xbase + kx)) * CI + ci0];
  };

  for (int s = 0; s < NSTG; ++s) {
    const int kb0  = s * KCH;
    const int rows = (KTOT - kb0) < KCH ? (KTOT - kb0) : KCH;
    if (s + 1 < NSTG) {
      if (wv == 0) {
        int nrows = (KTOT - kb0 - KCH) < KCH ? (KTOT - kb0 - KCH) : KCH;
        tdm_load_panel(gpanel + (size_t)(kb0 + KCH) * Co,
                       (unsigned)(uintptr_t)&s_w[(s + 1) & 1][0], nrows, wco, Co);
      }
      tdm_wait(1);                                // stage s complete
    } else {
      tdm_wait(0);
    }
#if !HAVE_TDM
    // fallback: coalesced cooperative copy of this stage's panel
    for (int i = tid; i < (rows << wcosh) >> 3; i += 256) {
      int e = i << 3;
      int c = e & (wco - 1), rr = e >> wcosh;
      *(uint4*)&s_w[s & 1][e] =
          *(const uint4*)&gpanel[(size_t)(kb0 + rr) * Co + c];
    }
#endif
    __syncthreads();                              // panel (and patch) visible

    const h16* wp = s_w[s & 1];
    const int nf = rows >> 5;
    for (int c = 0; c < nf; ++c) {
      const int kk = kb0 + (c << 5);
      VU a;
      a.q[0] = loadA(kk);
      a.q[1] = loadA(kk + 16);
      v16h bfr;
      for (int s2 = 0; s2 < 16; ++s2)
        bfr[s2] = wp[(((c << 5) + (s2 < 8 ? khalf + s2 : 16 + khalf + s2 - 8)) << wcosh) + (ni << 4) + m];
      acc = __builtin_amdgcn_wmma_f32_16x16x32_f16(
          false, a.v, false, bfr, (short)0, acc, false, false);
    }
    if (rows & 16) {                              // 16-K tail, upper half 0
      const int kk = kb0 + (nf << 5);
      VU a;
      a.q[0] = loadA(kk);
      a.q[1] = make_uint4(0u, 0u, 0u, 0u);
      v16h bfr;
      for (int s2 = 0; s2 < 16; ++s2)
        bfr[s2] = s2 < 8 ? wp[(((nf << 5) + khalf + s2) << wcosh) + (ni << 4) + m] : (h16)0;
      acc = __builtin_amdgcn_wmma_f32_16x16x32_f16(
          false, a.v, false, bfr, (short)0, acc, false, false);
    }
    __syncthreads();                              // before buffer reuse
  }

  if (active) {
    const int n    = lane & 15;
    const int mhi  = (lane >> 4) << 3;            // C: m = r + 8*(lane>=16)
    const int co   = co0 + n;
    const int cosh = (Co == 16 ? 4 : Co == 32 ? 5 : Co == 64 ? 6 : 7);
    const float bs = bias[co];
    #pragma unroll
    for (int r = 0; r < 8; ++r) {
      int x = x0 + mi * 16 + (r + mhi);
      size_t oidx = ((size_t)((((b << 8) + y) << 8) + x) << cosh) + co;
      float v = acc[r] + bs;
      if (skip) v += (float)skip[oidx];
      if (relu) v = v > 0.f ? v : 0.f;
      out[oidx] = (h16)v;
    }
  }
}

// ---------------------------------------------------------------------------
// Correlation (169 displacements x 64 ch) + softmax + attention-gather.
// The would-be GEMM's B operand depends on both pixel (m) and displacement
// (n), so it cannot map to WMMA; at 5.7 GFLOP it runs on VALU.
// ---------------------------------------------------------------------------
__global__ __launch_bounds__(256) void k_corr_softmax(
    const h16* __restrict__ ft, const h16* __restrict__ fr,
    const float* __restrict__ img, float* __restrict__ out)
{
  __shared__ h16   s_fr[13 * 20 * 64];   // 33280 B
  __shared__ h16   s_ft[8 * 64];
  __shared__ float s_img[13 * 20];

  const int xg = blockIdx.x & 31;        // W/8 = 32
  const int by = blockIdx.x >> 5;
  const int y  = by & 255, b = by >> 8;
  const int x0 = xg * 8;
  const int tid = threadIdx.x;

  for (int i = tid; i < 13 * 20 * 8; i += 256) {
    int ch = (i & 7) << 3, rc = i >> 3;
    int c = rc % 20, r = rc / 20;
    int yy = y + r - 6, xx = x0 + c - 6;
    uint4 val = make_uint4(0u, 0u, 0u, 0u);
    if ((unsigned)yy < 256u && (unsigned)xx < 256u)
      val = *(const uint4*)&fr[((((size_t)b << 8) + yy) * 256 + xx) * 64 + ch];
    *(uint4*)&s_fr[(r * 20 + c) * 64 + ch] = val;
  }
  if (tid < 64) {
    int px = tid >> 3, ch = (tid & 7) << 3;
    *(uint4*)&s_ft[px * 64 + ch] =
        *(const uint4*)&ft[((((size_t)b << 8) + y) * 256 + (x0 + px)) * 64 + ch];
  }
  for (int i = tid; i < 260; i += 256) {
    int c = i % 20, r = i / 20;
    int yy = y + r - 6, xx = x0 + c - 6;
    float v = 0.f;
    if ((unsigned)yy < 256u && (unsigned)xx < 256u)
      v = img[(((size_t)b << 8) + yy) * 256 + xx];
    s_img[r * 20 + c] = v;
  }
  __syncthreads();

  const int w    = tid >> 5;
  const int lane = tid & 31;

  float corr[6], iv[6];
  bool  ok[6];
  #pragma unroll
  for (int j = 0; j < 6; ++j) {
    int d = lane + 32 * j;
    bool valid = d < 169;
    int dd = valid ? d : 0;
    int dy = dd / 13, dx = dd - 13 * dy;
    float acc = 0.f;
    const h16* tp = &s_ft[w * 64];
    const h16* fp = &s_fr[(dy * 20 + (w + dx)) * 64];
    #pragma unroll
    for (int t = 0; t < 16; ++t) {
      h4 fa = *(const h4*)&tp[t * 4];
      h4 fb = *(const h4*)&fp[t * 4];
      acc += (float)fa.x * (float)fb.x + (float)fa.y * (float)fb.y +
             (float)fa.z * (float)fb.z + (float)fa.w * (float)fb.w;
    }
    corr[j] = valid ? acc : -3.0e38f;
    iv[j]   = s_img[dy * 20 + (w + dx)];
    ok[j]   = valid;
  }

  float mx = corr[0];
  #pragma unroll
  for (int j = 1; j < 6; ++j) mx = fmaxf(mx, corr[j]);
  for (int o = 16; o; o >>= 1) mx = fmaxf(mx, __shfl_xor(mx, o, 32));

  float es = 0.f, ws = 0.f;
  #pragma unroll
  for (int j = 0; j < 6; ++j)
    if (ok[j]) { float e = __expf(corr[j] - mx); es += e; ws += e * iv[j]; }
  for (int o = 16; o; o >>= 1) {
    es += __shfl_xor(es, o, 32);
    ws += __shfl_xor(ws, o, 32);
  }
  if (lane == 0) out[(((size_t)b << 8) + y) * 256 + x0 + w] = ws / es;
}

// ---------------------------------------------------------------------------
// Host orchestration
// ---------------------------------------------------------------------------
extern "C" void kernel_launch(void* const* d_in, const int* in_sizes, int n_in,
                              void* d_out, int out_size, void* d_ws, size_t ws_size,
                              hipStream_t stream)
{
  (void)in_sizes; (void)n_in; (void)out_size; (void)ws_size;
  const int B = BATCH, H = HW256, W = HW256, N = B * H * W;

  // ---- param leaves in JAX pytree (sorted-key) order ----
  static const int CIs[8] = {16, 16, 16, 32, 32, 64, 64, 128};
  static const int COs[8] = {16, 16, 32, 32, 64, 64, 128, 128};
  static const int SCs[8] = {0, 0, 1, 0, 1, 0, 1, 0};

  struct BN { const float *b, *g, *m, *v; };
  int idx = 0;
  auto nf = [&]() { return (const float*)d_in[idx++]; };
  auto getbn = [&]() { BN r; r.b = nf(); r.g = nf(); r.m = nf(); r.v = nf(); return r; };

  BN bn1[8], bn2[8], scbn[8];
  const float *c1w[8], *c2w[8], *scw[8];
  for (int i = 0; i < 8; ++i) {
    bn1[i] = getbn(); bn2[i] = getbn();
    c1w[i] = nf(); c2w[i] = nf();
    if (SCs[i]) { scw[i] = nf(); scbn[i] = getbn(); } else scw[i] = nullptr;
  }
  BN cbn = getbn();
  const float* conv1w = nf();
  const float* postb  = nf();
  const float* postw  = nf();
  const float* slice1 = nf();   // reference image (windows source)
  const float* s1in   = nf();   // slice1_input -> fr
  const float* s2in   = nf();   // slice2_input -> ft

  // ---- bump allocator over workspace ----
  char* wsp = (char*)d_ws;
  size_t off = 0;
  auto alloc = [&](size_t bytes) -> void* {
    off = (off + 255) & ~(size_t)255;
    void* p = wsp + off;
    off += bytes;
    return p;
  };

  h16 *w16_c1[8], *w16_c2[8], *w16_sc[8], *w16_conv1, *w16_post;
  float *b_c1[8], *b_c2[8], *b_sc[8], *b_conv1, *b_post;
  for (int i = 0; i < 8; ++i) {
    w16_c1[i] = (h16*)alloc((size_t)9 * CIs[i] * COs[i] * 2);
    b_c1[i]   = (float*)alloc((size_t)COs[i] * 4);
    w16_c2[i] = (h16*)alloc((size_t)9 * COs[i] * COs[i] * 2);
    b_c2[i]   = (float*)alloc((size_t)COs[i] * 4);
    if (SCs[i]) {
      w16_sc[i] = (h16*)alloc((size_t)CIs[i] * COs[i] * 2);
      b_sc[i]   = (float*)alloc((size_t)COs[i] * 4);
    }
  }
  w16_conv1 = (h16*)alloc(49 * 16 * 2);  b_conv1 = (float*)alloc(16 * 4);
  w16_post  = (h16*)alloc((size_t)9 * 128 * 64 * 2); b_post = (float*)alloc(64 * 4);

  h16* X  = (h16*)alloc((size_t)N * 128 * 2);
  h16* T  = (h16*)alloc((size_t)N * 128 * 2);
  h16* S  = (h16*)alloc((size_t)N * 128 * 2);
  h16* FR = (h16*)alloc((size_t)N * 64 * 2);
  h16* FT = (h16*)alloc((size_t)N * 64 * 2);

  // ---- weight prep (BN folding) ----
  auto prep = [&](const float* w, const BN* bn, const float* rawb,
                  int Co, int Ci, int KH, int KW, h16* w16, float* b32) {
    int tot = Co * Ci * KH * KW;
    k_prep_w<<<(tot + 255) / 256, 256, 0, stream>>>(
        w, bn ? bn->g : nullptr, bn ? bn->b : rawb, bn ? bn->m : nullptr,
        bn ? bn->v : nullptr, w16, b32, Co, Ci, KH, KW);
  };
  prep(conv1w, &cbn, nullptr, 16, 1, 7, 7, w16_conv1, b_conv1);
  for (int i = 0; i < 8; ++i) {
    prep(c1w[i], &bn1[i], nullptr, COs[i], CIs[i], 3, 3, w16_c1[i], b_c1[i]);
    prep(c2w[i], &bn2[i], nullptr, COs[i], COs[i], 3, 3, w16_c2[i], b_c2[i]);
    if (SCs[i]) prep(scw[i], &scbn[i], nullptr, COs[i], CIs[i], 1, 1, w16_sc[i], b_sc[i]);
  }
  prep(postw, nullptr, postb, 64, 128, 3, 3, w16_post, b_post);

  // ---- conv launchers (template dispatch on shape) ----
  auto conv3 = [&](const h16* in, const h16* w16, const float* b32,
                   const h16* skip, h16* outp, int Ci, int Co, int relu) {
    dim3 g(W / TW, B * H, (Co + 31) / 32);
    switch (Ci) {
      case 16:  k_conv_wmma<3,3,16> <<<g,256,0,stream>>>(in,w16,b32,skip,outp,Co,relu); break;
      case 32:  k_conv_wmma<3,3,32> <<<g,256,0,stream>>>(in,w16,b32,skip,outp,Co,relu); break;
      case 64:  k_conv_wmma<3,3,64> <<<g,256,0,stream>>>(in,w16,b32,skip,outp,Co,relu); break;
      default:  k_conv_wmma<3,3,128><<<g,256,0,stream>>>(in,w16,b32,skip,outp,Co,relu); break;
    }
  };
  auto conv1x1 = [&](const h16* in, const h16* w16, const float* b32,
                     h16* outp, int Ci, int Co) {
    dim3 g(W / TW, B * H, (Co + 31) / 32);
    switch (Ci) {
      case 16:  k_conv_wmma<1,1,16><<<g,256,0,stream>>>(in,w16,b32,nullptr,outp,Co,0); break;
      case 32:  k_conv_wmma<1,1,32><<<g,256,0,stream>>>(in,w16,b32,nullptr,outp,Co,0); break;
      default:  k_conv_wmma<1,1,64><<<g,256,0,stream>>>(in,w16,b32,nullptr,outp,Co,0); break;
    }
  };

  // ---- feature pyramid (shared weights, run twice) ----
  auto features = [&](const float* img_in, h16* feat) {
    k_conv_stem<<<B * (H / 16) * (W / 16), 256, 0, stream>>>(
        img_in, w16_conv1, b_conv1, X);
    h16 *cur = X, *other = S;
    for (int i = 0; i < 8; ++i) {
      conv3(cur, w16_c1[i], b_c1[i], nullptr, T, CIs[i], COs[i], 1);
      if (SCs[i]) {
        conv1x1(cur, w16_sc[i], b_sc[i], other, CIs[i], COs[i]);
        conv3(T, w16_c2[i], b_c2[i], /*skip=*/other, /*out=*/cur,
              COs[i], COs[i], 1);
      } else {
        conv3(T, w16_c2[i], b_c2[i], /*skip=*/cur, /*out=*/other,
              COs[i], COs[i], 1);
        h16* t2 = cur; cur = other; other = t2;
      }
    }
    conv3(cur, w16_post, b_post, nullptr, feat, 128, 64, 0);
  };
  features(s1in, FR);   // reference slice features
  features(s2in, FT);   // target slice features

  // ---- correlation softmax gather ----
  k_corr_softmax<<<B * H * (W / 8), 256, 0, stream>>>(
      FT, FR, slice1, (float*)d_out);
}
